// CrossViewSwapAttention_81673098101643
// MI455X (gfx1250) — matrix operand order repaired and checked by reading, not compiled, not run on hardware
//
#include <hip/hip_runtime.h>
#include <hip/hip_bf16.h>
#include <math.h>

// ---------------------------------------------------------------------------
// CDNA5 (gfx1250) implementation of CrossViewSwapAttention.
// All dense contractions go through v_wmma_f32_16x16x32_f16 (wave32 WMMA).
// Activations kept channel-major [C][token] so every linear layer is one
// generic WMMA GEMM.  Attention is fused per (window, head, 64-query chunk).
// B panels / V panels are staged through LDS (bank-padded) so the WMMA
// operand feeds are contiguous ds_load_b128, not redundant global gathers.
// ---------------------------------------------------------------------------

typedef __attribute__((ext_vector_type(16))) _Float16 v16h;
typedef __attribute__((ext_vector_type(8)))  float    v8f;

__device__ __forceinline__ v8f wmma16x16x32(v16h a, v16h b, v8f c) {
  // 8 args: (neg_a, A, neg_b, B, c_mod, C, reuse_a, reuse_b)
  return __builtin_amdgcn_wmma_f32_16x16x32_f16(false, a, false, b, (short)0, c,
                                                false, false);
}

// A-fragment K index for 16-bit A 16x32 layout (matches B order):
//   lane<16 : K = 0..15 ; lane>=16 : K = 16..31
__device__ __forceinline__ int a_kidx(int i, int group) {
  return (group ? 16 : 0) + i;
}
// B-fragment K index for 16-bit B 32x16 layout:
//   lane<16 : K = 0..15 ; lane>=16 : K = 16..31 (2 halves per VGPR)
__device__ __forceinline__ int b_kidx(int i, int group) {
  return (group ? 16 : 0) + i;
}

// ---------------------------------------------------------------------------
// Generic GEMM:  C[M,N] = act( A[M,K] @ B[K,N] + bias[M] (+ res[M,N]) )
// A: f32 row-major weights.  B,C,res: f32 channel-major [rows][N].
// blockDim = 256 (8 waves); wave w computes 16x16 tile rows m0=by*128+w*16,
// cols n0 = bx*16.  The shared K x 16 B-panel is staged once into LDS as f16
// with +8 half padding per column (breaks 64-bank conflicts), then every
// wave reads fragments as contiguous ds_load_b128.  K-loop fully unrolled
// (template) -> K/32 static v_wmma.  ACT: 0 = none, 1 = exact GELU.
// ---------------------------------------------------------------------------
template <int K, int ACT>
__global__ __launch_bounds__(256) void gemm_wmma(
    const float* __restrict__ A, const float* __restrict__ B,
    float* __restrict__ C, const float* __restrict__ bias,
    const float* __restrict__ res, int N) {
  constexpr int KP = K + 8;                 // padded K stride (halves)
  __shared__ _Float16 lB[16 * KP];          // B panel, layout [n][k]

  const int tid   = threadIdx.x;
  const int lane  = tid & 31;
  const int wave  = tid >> 5;
  const int group = lane >> 4;
  const int ln16  = lane & 15;
  const int n0    = blockIdx.x * 16;
  const int m0    = blockIdx.y * 128 + wave * 16;

  if (res)  // next-use hint for the residual stream (global_prefetch_b8)
    __builtin_prefetch(res + (size_t)(m0 + ln16) * N + n0, 0, 1);

  // ---- stage B panel: K*16 elements, 256 threads -> K/16 each, coalesced
#pragma unroll
  for (int it = 0; it < (K * 16) / 256; ++it) {
    const int e = tid + it * 256;           // e = k*16 + n
    const int n = e & 15, k = e >> 4;
    lB[n * KP + k] = (_Float16)B[(size_t)k * N + n0 + n];
  }
  __syncthreads();

  v8f acc;
#pragma unroll
  for (int r = 0; r < 8; ++r)
    acc[r] = bias ? bias[m0 + r + 8 * group] : 0.0f;

  const float*   Arow = A + (size_t)(m0 + ln16) * K;
  const _Float16* Bln = lB + ln16 * KP;
#pragma unroll
  for (int k0 = 0; k0 < K; k0 += 32) {
    v16h a, b;
#pragma unroll
    for (int i = 0; i < 16; ++i) {
      a[i] = (_Float16)Arow[k0 + a_kidx(i, group)];
      b[i] = Bln[k0 + b_kidx(i, group)];    // contiguous 16 halves / group
    }
    acc = wmma16x16x32(a, b, acc);
  }

#pragma unroll
  for (int r = 0; r < 8; ++r) {
    const int m = m0 + r + 8 * group;
    const size_t idx = (size_t)m * N + n0 + ln16;
    float v = acc[r];
    if (res) v += res[idx];
    if (ACT == 1) v = 0.5f * v * (1.0f + erff(v * 0.70710678118654752f));
    C[idx] = v;
  }
}

// ---------------------------------------------------------------------------
// Window token maps for the two attention variants ("swap" layer).
// Feature tokens: tk = n*4096 + h*64 + w  (64x64 per camera)
// Query tokens  : attn1: n*16384 + h*128 + w ; attn2: h*128 + w
// ---------------------------------------------------------------------------
__device__ __forceinline__ int qTokenF(int l, int j, int mode) {
  const int qx = l >> 3, qy = l & 7;
  if (mode == 1) {
    const int n = j >> 8, rem = j & 255;
    return (n * 128 + qx * 16 + (rem >> 4)) * 128 + qy * 16 + (rem & 15);
  }
  return (qx * 16 + (j >> 4)) * 128 + qy * 16 + (j & 15);
}
__device__ __forceinline__ int kTokenF(int l, int j, int mode) {
  const int n = j >> 6, rem = j & 63;
  int h, w;
  if (mode == 1) { h = (l >> 3) * 8 + (rem >> 3);  w = (l & 7) * 8 + (rem & 7); }
  else           { h = (rem >> 3) * 8 + (l >> 3);  w = (rem & 7) * 8 + (l & 7); }
  return (n * 64 + h) * 64 + w;
}

// ---------------------------------------------------------------------------
// Fused attention: per (window l, head h, chunk of 64 queries).
// S = softmax(scale * Q Kt) via WMMA, P stored f16 in LDS, O = P V via WMMA.
// Key-token indices and the 32x256 V head-panel are staged in LDS so the
// P@V inner loop is pure ds_load_b128 + v_wmma.
// blockDim = 128 (4 waves, 16 queries each).  mode 1 = attn1, 2 = attn2.
// ---------------------------------------------------------------------------
__global__ __launch_bounds__(128) void attn_wmma(
    const float* __restrict__ Qp, const float* __restrict__ Kp,
    const float* __restrict__ Vp, float* __restrict__ Op, int mode) {
  __shared__ _Float16 lP[64 * 264];   // scores / probabilities (padded rows)
  __shared__ _Float16 lV[32 * 264];   // V head panel, [c][k] padded
  __shared__ int      lKt[256];       // per-window key token indices
  __shared__ float    red[128];

  const int l = blockIdx.x, h = blockIdx.y, chunk = blockIdx.z;
  const int tid   = threadIdx.x;
  const int lane  = tid & 31, wave = tid >> 5;
  const int group = lane >> 4, ln16 = lane & 15;
  const int NQ = (mode == 1) ? 65536 : 16384;   // query token count
  const int NK = 16384;                         // key token count
  const int qbase = chunk * 64 + wave * 16;
  const float scale = 0.17677669529663687f;     // 32^-0.5 folded into Q

  // ---- key token indices (removes index math from the hot loops)
  for (int e = tid; e < 256; e += 128) lKt[e] = kTokenF(l, e, mode);
  __syncthreads();

  // ---- stage V head panel: lV[c][k] = V'[key k][channel h*32+c]
  for (int e = tid; e < 32 * 256; e += 128) {
    const int c = e >> 8, k = e & 255;
    lV[c * 264 + k] = (_Float16)Vp[(size_t)(h * 32 + c) * NK + lKt[k]];
  }

  { // ---- S = scale * Q @ K^T  (contraction dim = dh = 32, one WMMA/tile)
    v16h a;
    const int qt = qTokenF(l, qbase + ln16, mode);
#pragma unroll
    for (int i = 0; i < 16; ++i)
      a[i] = (_Float16)(scale * Qp[(size_t)(h * 32 + a_kidx(i, group)) * NQ + qt]);
#pragma unroll 4
    for (int nt = 0; nt < 16; ++nt) {
      const int kt = lKt[nt * 16 + ln16];
      v16h b;
#pragma unroll
      for (int i = 0; i < 16; ++i)
        b[i] = (_Float16)Kp[(size_t)(h * 32 + b_kidx(i, group)) * NK + kt];
      v8f acc;
#pragma unroll
      for (int r = 0; r < 8; ++r) acc[r] = 0.0f;
      acc = wmma16x16x32(a, b, acc);
#pragma unroll
      for (int r = 0; r < 8; ++r)
        lP[(wave * 16 + r + 8 * group) * 264 + nt * 16 + ln16] = (_Float16)acc[r];
    }
  }
  __syncthreads();

  { // ---- row softmax over 256 keys (thread owns half a row, f32 math)
    const int r = tid >> 1, s = tid & 1;
    _Float16* row = lP + r * 264 + s * 128;
    float mx = -1e30f;
    for (int c = 0; c < 128; ++c) mx = fmaxf(mx, (float)row[c]);
    red[tid] = mx;
    __syncthreads();
    mx = fmaxf(red[r * 2], red[r * 2 + 1]);
    __syncthreads();
    float sum = 0.0f;
    for (int c = 0; c < 128; ++c) {
      float e = __expf((float)row[c] - mx);
      row[c] = (_Float16)e;
      sum += e;
    }
    red[tid] = sum;
    __syncthreads();
    const float inv = 1.0f / (red[r * 2] + red[r * 2 + 1]);
    for (int c = 0; c < 128; ++c) row[c] = (_Float16)((float)row[c] * inv);
  }
  __syncthreads();

  { // ---- O = P @ V  (M=16 queries, N=32 channels, K=256 keys, all LDS)
    const _Float16* Prow = lP + (wave * 16 + ln16) * 264;
#pragma unroll
    for (int nt = 0; nt < 2; ++nt) {
      const _Float16* Vcol = lV + (nt * 16 + ln16) * 264;
      v8f acc;
#pragma unroll
      for (int r = 0; r < 8; ++r) acc[r] = 0.0f;
#pragma unroll
      for (int k0 = 0; k0 < 256; k0 += 32) {
        v16h a, b;
#pragma unroll
        for (int i = 0; i < 16; ++i) {
          a[i] = Prow[k0 + a_kidx(i, group)];
          b[i] = Vcol[k0 + b_kidx(i, group)];  // contiguous ds_load_b128
        }
        acc = wmma16x16x32(a, b, acc);
      }
#pragma unroll
      for (int r = 0; r < 8; ++r) {
        const int qt = qTokenF(l, qbase + r + 8 * group, mode);
        Op[(size_t)(h * 32 + nt * 16 + ln16) * NQ + qt] = acc[r];
      }
    }
  }
}

// ---------------------------------------------------------------------------
// LayerNorm over 128 channels, channel-major [128][Ntok]; one wave per token.
// ---------------------------------------------------------------------------
__global__ __launch_bounds__(128) void ln_kernel(
    const float* __restrict__ X, float* __restrict__ Y,
    const float* __restrict__ g, const float* __restrict__ b, int Ntok) {
  const int t    = blockIdx.x * 4 + (threadIdx.x >> 5);
  const int lane = threadIdx.x & 31;
  float v[4], s = 0.0f;
#pragma unroll
  for (int i = 0; i < 4; ++i) {
    v[i] = X[(size_t)(lane + 32 * i) * Ntok + t];
    s += v[i];
  }
#pragma unroll
  for (int off = 16; off > 0; off >>= 1) s += __shfl_xor(s, off, 32);
  const float mu = s * (1.0f / 128.0f);
  float q = 0.0f;
#pragma unroll
  for (int i = 0; i < 4; ++i) { float d = v[i] - mu; q += d * d; }
#pragma unroll
  for (int off = 16; off > 0; off >>= 1) q += __shfl_xor(q, off, 32);
  const float inv = rsqrtf(q * (1.0f / 128.0f) + 1e-5f);
#pragma unroll
  for (int i = 0; i < 4; ++i) {
    const int c = lane + 32 * i;
    Y[(size_t)c * Ntok + t] = (v[i] - mu) * inv * g[c] + b[c];
  }
}

// ---------------------------------------------------------------------------
// eval-mode BatchNorm + ReLU, transposing feature [n][c][p] -> act [c][n*4096+p]
// ---------------------------------------------------------------------------
__global__ void bnrelu_kernel(const float* __restrict__ F,
                              const float* __restrict__ g,
                              const float* __restrict__ b,
                              float* __restrict__ A) {
  const int i = blockIdx.x * 256 + threadIdx.x;   // over 4*128*4096
  const int n = i >> 19, c = (i >> 12) & 127, p = i & 4095;
  const float v = F[i] * (g[c] * rsqrtf(1.0f + 1e-5f)) + b[c];
  A[(size_t)c * 16384 + n * 4096 + p] = fmaxf(v, 0.0f);
}

// ---------------------------------------------------------------------------
// img_embed (normalized) -> Key residual buffer, channel-major [c][n*4096+p]
// ---------------------------------------------------------------------------
__global__ __launch_bounds__(128) void embed_img_kernel(
    const float* __restrict__ Iinv, const float* __restrict__ Einv,
    const float* __restrict__ Wimg, const float* __restrict__ Wcam,
    float* __restrict__ Key) {
  __shared__ float sred[128];
  const int n = blockIdx.x >> 12, p = blockIdx.x & 4095;
  const int i = p >> 6, j = p & 63;
  const float px = (j * (1.0f / 63.0f)) * 704.0f;   // IMG_W
  const float py = (i * (1.0f / 63.0f)) * 256.0f;   // IMG_H
  const float* I = Iinv + n * 9;
  float cam[4];
  cam[0] = I[0] * px + I[1] * py + I[2];
  cam[1] = I[3] * px + I[4] * py + I[5];
  cam[2] = I[6] * px + I[7] * py + I[8];
  cam[3] = 1.0f;
  const float* E = Einv + n * 16;
  float dv[4];
#pragma unroll
  for (int r = 0; r < 4; ++r)
    dv[r] = E[r * 4 + 0] * cam[0] + E[r * 4 + 1] * cam[1] +
            E[r * 4 + 2] * cam[2] + E[r * 4 + 3] * cam[3];
  const int o = threadIdx.x;
  float e = 0.0f, ce = 0.0f;
#pragma unroll
  for (int c = 0; c < 4; ++c) {
    e  += Wimg[o * 4 + c] * dv[c];
    ce += Wcam[o * 4 + c] * E[c * 4 + 3];
  }
  e -= ce;
  sred[o] = e * e;
  __syncthreads();
  for (int s = 64; s > 0; s >>= 1) {
    if (o < s) sred[o] += sred[o + s];
    __syncthreads();
  }
  const float nrm = sqrtf(sred[0]);
  Key[(size_t)o * 16384 + n * 4096 + p] = e / (nrm + 1e-7f);
}

// ---------------------------------------------------------------------------
// query = normalized(bev_embed) + x ; channel-major [c][n*16384 + h*128 + w]
// ---------------------------------------------------------------------------
__global__ __launch_bounds__(128) void embed_query_kernel(
    const float* __restrict__ grid, const float* __restrict__ Einv,
    const float* __restrict__ Wbev, const float* __restrict__ bbev,
    const float* __restrict__ Wcam, const float* __restrict__ x,
    float* __restrict__ Q) {
  __shared__ float sred[128];
  const int n = blockIdx.x >> 14, t = blockIdx.x & 16383;
  const int o = threadIdx.x;
  const float g0 = grid[t], g1 = grid[16384 + t];
  float e = Wbev[o * 2] * g0 + Wbev[o * 2 + 1] * g1 + bbev[o];
  const float* E = Einv + n * 16;
  float ce = 0.0f;
#pragma unroll
  for (int c = 0; c < 4; ++c) ce += Wcam[o * 4 + c] * E[c * 4 + 3];
  e -= ce;
  sred[o] = e * e;
  __syncthreads();
  for (int s = 64; s > 0; s >>= 1) {
    if (o < s) sred[o] += sred[o + s];
    __syncthreads();
  }
  const float nrm = sqrtf(sred[0]);
  Q[(size_t)o * 65536 + n * 16384 + t] = e / (nrm + 1e-7f) + x[(size_t)o * 16384 + t];
}

// ---------------------------------------------------------------------------
// Z[c][t] = skip[c][t] + mean_n Zo[c][n*16384 + t]
// ---------------------------------------------------------------------------
__global__ void mean_skip_kernel(const float* __restrict__ Zo,
                                 const float* __restrict__ skip,
                                 float* __restrict__ Z, int ncam) {
  const int i = blockIdx.x * 256 + threadIdx.x;   // over 128*16384
  const int c = i >> 14, t = i & 16383;
  float s = 0.0f;
  for (int n = 0; n < ncam; ++n)
    s += Zo[(size_t)c * ncam * 16384 + (size_t)n * 16384 + t];
  Z[i] = skip[i] + s / (float)ncam;
}

// ---------------------------------------------------------------------------
extern "C" void kernel_launch(void* const* d_in, const int* in_sizes, int n_in,
                              void* d_out, int out_size, void* d_ws,
                              size_t ws_size, hipStream_t stream) {
  (void)in_sizes; (void)n_in; (void)out_size; (void)ws_size;
  auto F = [&](int i) { return (const float*)d_in[i]; };
  // input order (setup_inputs dict order, params dict flattened in order):
  // 0 x, 1 feature, 2 I_inv, 3 E_inv, 4 grid,
  // 5 W_cam, 6 W_img, 7 W_bev, 8 b_bev, 9 kbn_g, 10 kbn_b, 11 W_kproj,
  // 12 vbn_g, 13 vbn_b, 14 W_vlin,
  // 15..20 a1 qn/kn/vn g,b; 21..24 a1 Wq,Wk,Wv,Wo; 25..28 a1 bq,bk,bv,bo
  // 29..34 a2 qn/kn/vn g,b; 35..38 a2 Wq,Wk,Wv,Wo; 39..42 a2 bq,bk,bv,bo
  // 43..46 mlp1 W1,b1,W2,b2; 47..50 mlp2 W1,b1,W2,b2;
  // 51..56 pre1_g,pre1_b,pre2_g,pre2_b,post_g,post_b; 57 index
  const float* x = F(0);
  float* ws = (float*)d_ws;
  const size_t U = 2097152;              // 128*16384 floats
  float* act  = ws + 0 * U;
  float* key  = ws + 1 * U;
  float* val  = ws + 2 * U;
  float* z    = ws + 3 * U;              // attn1 output / z2
  float* z1   = ws + 4 * U;              // post-mlp1 / z3
  float* Kp   = ws + 5 * U;
  float* Vp   = ws + 6 * U;
  float* h1   = ws + 7 * U;              // 2U (MLP hidden, 256 ch)
  float* qbuf = ws + 9 * U;              // 4U (query, reused as Op)
  float* tmp  = ws + 13 * U;             // 4U (LN out, reused as Zo)
  float* Qp   = ws + 17 * U;             // 4U

  // ---- key / value feature projections (1x1 conv == GEMM, N = 4*64*64)
  bnrelu_kernel<<<8192, 256, 0, stream>>>(F(1), F(9), F(10), act);
  embed_img_kernel<<<16384, 128, 0, stream>>>(F(2), F(3), F(6), F(5), key);
  gemm_wmma<128, 0><<<dim3(1024, 1), 256, 0, stream>>>(F(11), act, key, nullptr, key, 16384);
  bnrelu_kernel<<<8192, 256, 0, stream>>>(F(1), F(12), F(13), act);
  gemm_wmma<128, 0><<<dim3(1024, 1), 256, 0, stream>>>(F(14), act, val, nullptr, nullptr, 16384);

  // ---- query = query_pos + x (per camera)
  embed_query_kernel<<<65536, 128, 0, stream>>>(F(4), F(3), F(7), F(8), F(5), x, qbuf);

  // ---- attention 1: LN + QKV projections (all windows fused as one GEMM)
  ln_kernel<<<16384, 128, 0, stream>>>(qbuf, tmp, F(15), F(16), 65536);
  gemm_wmma<128, 0><<<dim3(4096, 1), 256, 0, stream>>>(F(21), tmp, Qp, F(25), nullptr, 65536);
  ln_kernel<<<4096, 128, 0, stream>>>(key, tmp, F(17), F(18), 16384);
  gemm_wmma<128, 0><<<dim3(1024, 1), 256, 0, stream>>>(F(22), tmp, Kp, F(26), nullptr, 16384);
  ln_kernel<<<4096, 128, 0, stream>>>(val, tmp, F(19), F(20), 16384);
  gemm_wmma<128, 0><<<dim3(1024, 1), 256, 0, stream>>>(F(23), tmp, Vp, F(27), nullptr, 16384);

  attn_wmma<<<dim3(64, 4, 16), 128, 0, stream>>>(Qp, Kp, Vp, qbuf, 1);
  gemm_wmma<128, 0><<<dim3(4096, 1), 256, 0, stream>>>(F(24), qbuf, tmp, F(28), nullptr, 65536);
  mean_skip_kernel<<<8192, 256, 0, stream>>>(tmp, x, z, 4);

  // ---- MLP 1 (GELU)
  ln_kernel<<<4096, 128, 0, stream>>>(z, tmp, F(51), F(52), 16384);
  gemm_wmma<128, 1><<<dim3(1024, 2), 256, 0, stream>>>(F(43), tmp, h1, F(44), nullptr, 16384);
  gemm_wmma<256, 0><<<dim3(1024, 1), 256, 0, stream>>>(F(45), h1, z1, F(46), z, 16384);

  // ---- attention 2 (swapped window/grid axes; query broadcast over cams,
  //      so per-camera outputs are identical -> compute once, skip the mean)
  ln_kernel<<<4096, 128, 0, stream>>>(z1, tmp, F(29), F(30), 16384);
  gemm_wmma<128, 0><<<dim3(1024, 1), 256, 0, stream>>>(F(35), tmp, Qp, F(39), nullptr, 16384);
  ln_kernel<<<4096, 128, 0, stream>>>(key, tmp, F(31), F(32), 16384);
  gemm_wmma<128, 0><<<dim3(1024, 1), 256, 0, stream>>>(F(36), tmp, Kp, F(40), nullptr, 16384);
  ln_kernel<<<4096, 128, 0, stream>>>(val, tmp, F(33), F(34), 16384);
  gemm_wmma<128, 0><<<dim3(1024, 1), 256, 0, stream>>>(F(37), tmp, Vp, F(41), nullptr, 16384);

  attn_wmma<<<dim3(64, 4, 4), 128, 0, stream>>>(Qp, Kp, Vp, qbuf, 2);
  gemm_wmma<128, 0><<<dim3(1024, 1), 256, 0, stream>>>(F(38), qbuf, tmp, F(42), nullptr, 16384);
  mean_skip_kernel<<<8192, 256, 0, stream>>>(tmp, z1, z, 1);    // z <- z2

  // ---- MLP 2 (GELU) + final LayerNorm straight into d_out ([d][h][w])
  ln_kernel<<<4096, 128, 0, stream>>>(z, tmp, F(53), F(54), 16384);
  gemm_wmma<128, 1><<<dim3(1024, 2), 256, 0, stream>>>(F(47), tmp, h1, F(48), nullptr, 16384);
  gemm_wmma<256, 0><<<dim3(1024, 1), 256, 0, stream>>>(F(49), h1, z1, F(50), z, 16384);
  ln_kernel<<<4096, 128, 0, stream>>>(z1, (float*)d_out, F(55), F(56), 16384);
}
